// Phi4FlashAttention_56650618635025
// MI455X (gfx1250) — matrix operand order repaired and checked by reading, not compiled, not run on hardware
//
#include <hip/hip_runtime.h>
#include <hip/hip_bf16.h>
#include <math.h>

// ---------------------------------------------------------------------------
// Phi4 differential flash attention for MI455X (gfx1250), wave32 + WMMA bf16
// ---------------------------------------------------------------------------

typedef __attribute__((ext_vector_type(16))) __bf16 v16bf;
typedef __attribute__((ext_vector_type(8)))  __bf16 bf16x8;
typedef __attribute__((ext_vector_type(4)))  __bf16 bf16x4;
typedef __attribute__((ext_vector_type(8)))  float  v8f;

#define S_LEN      2048
#define HDIM       2560
#define QKV_N      3072   // 40*64 + 2*4*64
#define NHP        20     // NH/2 head-pairs
#define HEADD      64
#define LAMBDA_INIT 0.796341952f
#define ATTN_SCALE  0.125f

// native converts -> v_cvt_pk_bf16_f32 pairs
__device__ __forceinline__ bf16x4 cvt4(float4 v) {
  bf16x4 r = { (__bf16)v.x, (__bf16)v.y, (__bf16)v.z, (__bf16)v.w };
  return r;
}
__device__ __forceinline__ bf16x4 cvt4s(float4 v, float s) {
  bf16x4 r = { (__bf16)(v.x * s), (__bf16)(v.y * s),
               (__bf16)(v.z * s), (__bf16)(v.w * s) };
  return r;
}

// A-fragment (16x32 bf16, row-major tile, row stride ld elements).
// ISA table: lanes 0-15 -> row=lane, K={0..7,16..23}; lanes 16-31 -> K={8..15,24..31}
__device__ __forceinline__ v16bf load_frag_a(const __bf16* tile, int ld) {
  const int lane = threadIdx.x & 31;
  const int row  = lane & 15;
  const int k0   = (lane >> 4) << 3;          // 0 or 8
  const __bf16* p = tile + row * ld;
  bf16x8 lo = *(const bf16x8*)(p + k0);
  bf16x8 hi = *(const bf16x8*)(p + 16 + k0);
  v16bf f;
#pragma unroll
  for (int i = 0; i < 8; ++i) { f[i] = lo[i]; f[8 + i] = hi[i]; }
  return f;
}

// B-fragment (32x16 bf16). Tile stored as B^T row-major: row = column n of B,
// contiguous along K. lanes 0-15 -> col=lane, K={0..7,16..23}; lanes 16-31 -> K={8..15,24..31}
__device__ __forceinline__ v16bf load_frag_b(const __bf16* tile, int ld) {
  const int lane = threadIdx.x & 31;
  const int col  = lane & 15;
  const int k0   = (lane >> 4) << 3;          // 0 or 8
  const __bf16* p = tile + col * ld + k0;
  bf16x8 lo = *(const bf16x8*)(p);
  bf16x8 hi = *(const bf16x8*)(p + 16);
  v16bf f;
#pragma unroll
  for (int i = 0; i < 8; ++i) { f[i] = lo[i]; f[8 + i] = hi[i]; }
  return f;
}

// ---------------------------------------------------------------------------
// Kernel 1/5: GEMM  C[M,N] = A[M,K] * W[N,K]^T + bias[N]   (fp32 in/out,
// bf16 WMMA internally).  128x128 block tile, 8 waves of 32x64, K-step 64
// (two WMMA k-chunks per barrier phase).
// ---------------------------------------------------------------------------
#define GLD 72   // padded LDS stride in bf16 (144B: 16B aligned, conflict-free)

__global__ __launch_bounds__(256)
void gemm_bf16w_kernel(const float* __restrict__ A, const float* __restrict__ W,
                       const float* __restrict__ bias, float* __restrict__ C,
                       int M, int N, int K) {
  __shared__ __align__(16) __bf16 sA[128 * GLD];
  __shared__ __align__(16) __bf16 sB[128 * GLD];
  const int tid  = threadIdx.x;
  const int m0   = blockIdx.y << 7;
  const int n0   = blockIdx.x << 7;
  const int w    = tid >> 5;
  const int wm   = w & 3;                 // 4 waves along M
  const int wn   = w >> 2;                // 2 waves along N
  const int lane = tid & 31;
  const int cn   = lane & 15;
  const int cm0  = (lane >> 4) << 3;

  v8f acc[2][4] = {};
  const int nK = K >> 6;                  // K-step 64

  for (int kb = 0; kb < nK; ++kb) {
    const int k0 = kb << 6;
    __syncthreads();
    // stage 128x64 fp32 -> bf16 for A and W (2048 float4 each / 256 threads)
#pragma unroll
    for (int i = 0; i < 8; ++i) {
      int idx = tid + (i << 8);
      int r   = idx >> 4;                 // row 0..127
      int c4  = (idx & 15) << 2;          // col 0,4,..,60
      float4 va = *(const float4*)(A + (size_t)(m0 + r) * K + k0 + c4);
      float4 vb = *(const float4*)(W + (size_t)(n0 + r) * K + k0 + c4);
      *(bf16x4*)(sA + r * GLD + c4) = cvt4(va);
      *(bf16x4*)(sB + r * GLD + c4) = cvt4(vb);
    }
    // prefetch next K tile into cache (global_prefetch_b8)
    if (kb + 1 < nK) {
      const int r    = tid >> 1;
      const int half = (tid & 1) << 5;
      __builtin_prefetch(A + (size_t)(m0 + r) * K + k0 + 64 + half, 0, 0);
      __builtin_prefetch(W + (size_t)(n0 + r) * K + k0 + 64 + half, 0, 0);
    }
    __syncthreads();

#pragma unroll
    for (int kk = 0; kk < 2; ++kk) {
      v16bf af0 = load_frag_a(sA + (wm * 32) * GLD + kk * 32, GLD);
      v16bf af1 = load_frag_a(sA + (wm * 32 + 16) * GLD + kk * 32, GLD);
#pragma unroll
      for (int ni = 0; ni < 4; ++ni) {
        v16bf bf = load_frag_b(sB + (wn * 64 + ni * 16) * GLD + kk * 32, GLD);
        acc[0][ni] = __builtin_amdgcn_wmma_f32_16x16x32_bf16(
            false, af0, false, bf, (short)0, acc[0][ni], false, false);
        acc[1][ni] = __builtin_amdgcn_wmma_f32_16x16x32_bf16(
            false, af1, false, bf, (short)0, acc[1][ni], false, false);
      }
    }
  }

#pragma unroll
  for (int mi = 0; mi < 2; ++mi) {
#pragma unroll
    for (int ni = 0; ni < 4; ++ni) {
      const int n     = n0 + wn * 64 + ni * 16 + cn;
      const int mbase = m0 + wm * 32 + mi * 16 + cm0;
      const float bv  = bias[n];
#pragma unroll
      for (int r = 0; r < 8; ++r)
        C[(size_t)(mbase + r) * N + n] = acc[mi][ni][r] + bv;
    }
  }
}

// ---------------------------------------------------------------------------
// Kernel 2/5: lambda_full = exp(sum(lq1*lk1)) - exp(sum(lq2*lk2)) + LAMBDA_INIT
// ---------------------------------------------------------------------------
__global__ __launch_bounds__(64)
void lambda_kernel(const float* lq1, const float* lk1,
                   const float* lq2, const float* lk2, float* out) {
  __shared__ float s1[64];
  __shared__ float s2[64];
  const int t = threadIdx.x;
  s1[t] = lq1[t] * lk1[t];
  s2[t] = lq2[t] * lk2[t];
  __syncthreads();
  if (t == 0) {
    float a = 0.f, b = 0.f;
    for (int i = 0; i < 64; ++i) { a += s1[i]; b += s2[i]; }
    out[0] = __expf(a) - __expf(b) + LAMBDA_INIT;
  }
}

// ---------------------------------------------------------------------------
// Kernel 3/5: causal flash attention for one (pair p, head-pair hp, 64-query
// block).  4 waves; each wave owns a 16-row query stripe.  Two outputs
// (P*V1, P*V2) share one softmax.  All matmuls via WMMA bf16.
// ---------------------------------------------------------------------------
#define ALD 72   // padded LDS stride (144B, 16B aligned)

__global__ __launch_bounds__(128)
void phi4_diff_attn_kernel(const float* __restrict__ qkv,
                           float* __restrict__ attn1,
                           float* __restrict__ attn2) {
  __shared__ __align__(16) __bf16 sQ [64 * ALD];
  __shared__ __align__(16) __bf16 sK [64 * ALD];
  __shared__ __align__(16) __bf16 sV1[64 * ALD];   // stored transposed [d][key]
  __shared__ __align__(16) __bf16 sV2[64 * ALD];   // stored transposed [d][key]
  __shared__ __align__(16) __bf16 sP [64 * ALD];

  const int tid  = threadIdx.x;
  const int w    = tid >> 5;
  const int lane = tid & 31;
  const int cn   = lane & 15;
  const int cm0  = (lane >> 4) << 3;

  const int qb  = blockIdx.x;        // 0..31   query block
  const int hp  = blockIdx.y;        // 0..19   head-pair
  const int p   = blockIdx.z;        // 0..1    differential pair
  const int kvp = hp / 10;           // GQA group (rep = 10)

  const int qoff  = (2 * hp + p) * HEADD;
  const int koff  = HDIM + (2 * kvp + p) * HEADD;
  const int voff1 = HDIM + 256 + (2 * kvp) * HEADD;
  const int voff2 = voff1 + HEADD;

  // stage Q (pre-scaled by 1/sqrt(D)) : 64x64 fp32 -> bf16
#pragma unroll
  for (int i = 0; i < 8; ++i) {
    int idx = tid + (i << 7);
    int row = idx >> 4;
    int c4  = (idx & 15) << 2;
    float4 q = *(const float4*)(qkv + (size_t)(qb * 64 + row) * QKV_N + qoff + c4);
    *(bf16x4*)(sQ + row * ALD + c4) = cvt4s(q, ATTN_SCALE);
  }

  v8f oa[4] = {};
  v8f ob[4] = {};
  float mrun[8], lrun[8];
#pragma unroll
  for (int r = 0; r < 8; ++r) { mrun[r] = -1e30f; lrun[r] = 0.f; }
  const int mrow = qb * 64 + w * 16 + cm0;

  for (int kb = 0; kb <= qb; ++kb) {
    __syncthreads();   // previous iteration's fragment reads done
    // stage K row-major, V1/V2 transposed
#pragma unroll
    for (int i = 0; i < 8; ++i) {
      int idx = tid + (i << 7);
      int key = idx >> 4;
      int c4  = (idx & 15) << 2;
      const float* rowp = qkv + (size_t)(kb * 64 + key) * QKV_N;
      float4 kk = *(const float4*)(rowp + koff + c4);
      *(bf16x4*)(sK + key * ALD + c4) = cvt4(kk);
      float4 v1 = *(const float4*)(rowp + voff1 + c4);
      sV1[(c4 + 0) * ALD + key] = (__bf16)v1.x;
      sV1[(c4 + 1) * ALD + key] = (__bf16)v1.y;
      sV1[(c4 + 2) * ALD + key] = (__bf16)v1.z;
      sV1[(c4 + 3) * ALD + key] = (__bf16)v1.w;
      float4 v2 = *(const float4*)(rowp + voff2 + c4);
      sV2[(c4 + 0) * ALD + key] = (__bf16)v2.x;
      sV2[(c4 + 1) * ALD + key] = (__bf16)v2.y;
      sV2[(c4 + 2) * ALD + key] = (__bf16)v2.z;
      sV2[(c4 + 3) * ALD + key] = (__bf16)v2.w;
    }
    __syncthreads();

    // S = Q * K^T  (already scaled)
    v8f sc[4] = {};
#pragma unroll
    for (int kk = 0; kk < 2; ++kk) {
      v16bf a = load_frag_a(sQ + (w * 16) * ALD + kk * 32, ALD);
#pragma unroll
      for (int ni = 0; ni < 4; ++ni) {
        v16bf b = load_frag_b(sK + (ni * 16) * ALD + kk * 32, ALD);
        sc[ni] = __builtin_amdgcn_wmma_f32_16x16x32_bf16(
            false, a, false, b, (short)0, sc[ni], false, false);
      }
    }

    // causal mask on the diagonal block
    if (kb == qb) {
#pragma unroll
      for (int ni = 0; ni < 4; ++ni)
#pragma unroll
        for (int r = 0; r < 8; ++r)
          if (kb * 64 + ni * 16 + cn > mrow + r) sc[ni][r] = -1e30f;
    }

    // online softmax
    float rmax[8];
#pragma unroll
    for (int r = 0; r < 8; ++r) {
      float v = sc[0][r];
      v = fmaxf(v, sc[1][r]); v = fmaxf(v, sc[2][r]); v = fmaxf(v, sc[3][r]);
      rmax[r] = v;
    }
#pragma unroll
    for (int off = 1; off < 16; off <<= 1)
#pragma unroll
      for (int r = 0; r < 8; ++r)
        rmax[r] = fmaxf(rmax[r], __shfl_xor(rmax[r], off, 32));

    float alpha[8], rsum[8];
#pragma unroll
    for (int r = 0; r < 8; ++r) {
      float mnew = fmaxf(mrun[r], rmax[r]);
      alpha[r] = __expf(mrun[r] - mnew);
      mrun[r]  = mnew;
      rsum[r]  = 0.f;
    }
#pragma unroll
    for (int ni = 0; ni < 4; ++ni)
#pragma unroll
      for (int r = 0; r < 8; ++r) {
        float pe = __expf(sc[ni][r] - mrun[r]);
        sc[ni][r] = pe;
        rsum[r] += pe;
      }
#pragma unroll
    for (int off = 1; off < 16; off <<= 1)
#pragma unroll
      for (int r = 0; r < 8; ++r)
        rsum[r] += __shfl_xor(rsum[r], off, 32);
#pragma unroll
    for (int r = 0; r < 8; ++r) lrun[r] = lrun[r] * alpha[r] + rsum[r];
#pragma unroll
    for (int j = 0; j < 4; ++j)
#pragma unroll
      for (int r = 0; r < 8; ++r) { oa[j][r] *= alpha[r]; ob[j][r] *= alpha[r]; }

    // P -> LDS (bf16), re-read as A fragment (same wave; DS is in-order)
#pragma unroll
    for (int ni = 0; ni < 4; ++ni)
#pragma unroll
      for (int r = 0; r < 8; ++r)
        sP[(w * 16 + cm0 + r) * ALD + ni * 16 + cn] = (__bf16)sc[ni][r];
    asm volatile("s_wait_dscnt 0" ::: "memory");

    // O += P * V1 , O2 += P * V2
#pragma unroll
    for (int kk = 0; kk < 2; ++kk) {
      v16bf a = load_frag_a(sP + (w * 16) * ALD + kk * 32, ALD);
#pragma unroll
      for (int j = 0; j < 4; ++j) {
        v16bf b1 = load_frag_b(sV1 + (j * 16) * ALD + kk * 32, ALD);
        oa[j] = __builtin_amdgcn_wmma_f32_16x16x32_bf16(
            false, a, false, b1, (short)0, oa[j], false, false);
        v16bf b2 = load_frag_b(sV2 + (j * 16) * ALD + kk * 32, ALD);
        ob[j] = __builtin_amdgcn_wmma_f32_16x16x32_bf16(
            false, a, false, b2, (short)0, ob[j], false, false);
      }
    }
  }

  float* outp = (p == 0) ? attn1 : attn2;   // [s][hp][128]
#pragma unroll
  for (int r = 0; r < 8; ++r) {
    const float invl = 1.0f / lrun[r];
    const size_t base = ((size_t)(mrow + r) * NHP + hp) * 128;
#pragma unroll
    for (int j = 0; j < 4; ++j) {
      const int d = j * 16 + cn;
      outp[base + d]      = oa[j][r] * invl;
      outp[base + 64 + d] = ob[j][r] * invl;
    }
  }
}

// ---------------------------------------------------------------------------
// Kernel 4/5: attn = rmsnorm(attn1 - lambda*attn2, subln_w) * (1-LAMBDA_INIT)
// one wave per 128-wide row; output laid out (s, 2560) = (s, hp*128 + c)
// ---------------------------------------------------------------------------
__global__ __launch_bounds__(256)
void diff_rms_kernel(const float* __restrict__ a1, const float* __restrict__ a2,
                     const float* __restrict__ wln, const float* __restrict__ lamp,
                     float* __restrict__ out) {
  const int wv   = threadIdx.x >> 5;
  const int lane = threadIdx.x & 31;
  const size_t row = (size_t)blockIdx.x * 8 + wv;    // < 2048*20
  const float lam = lamp[0];
  const float4 x1 = *(const float4*)(a1 + row * 128 + lane * 4);
  const float4 x2 = *(const float4*)(a2 + row * 128 + lane * 4);
  float y0 = x1.x - lam * x2.x, y1 = x1.y - lam * x2.y;
  float y2 = x1.z - lam * x2.z, y3 = x1.w - lam * x2.w;
  float ss = y0 * y0 + y1 * y1 + y2 * y2 + y3 * y3;
#pragma unroll
  for (int off = 1; off < 32; off <<= 1) ss += __shfl_xor(ss, off, 32);
  const float inv = rsqrtf(ss * (1.0f / 128.0f) + 1e-5f) * (1.0f - LAMBDA_INIT);
  const float4 g = *(const float4*)(wln + lane * 4);
  float4 o = { y0 * g.x * inv, y1 * g.y * inv, y2 * g.z * inv, y3 * g.w * inv };
  *(float4*)(out + row * 128 + lane * 4) = o;
}

// ---------------------------------------------------------------------------
extern "C" void kernel_launch(void* const* d_in, const int* in_sizes, int n_in,
                              void* d_out, int out_size, void* d_ws, size_t ws_size,
                              hipStream_t stream) {
  const float* hs     = (const float*)d_in[0];   // (2048, 2560)
  const float* Wqkv_w = (const float*)d_in[1];   // (3072, 2560)
  const float* Wqkv_b = (const float*)d_in[2];   // (3072,)
  const float* out_w  = (const float*)d_in[3];   // (2560, 2560)
  const float* out_b  = (const float*)d_in[4];   // (2560,)
  const float* lq1    = (const float*)d_in[5];
  const float* lk1    = (const float*)d_in[6];
  const float* lq2    = (const float*)d_in[7];
  const float* lk2    = (const float*)d_in[8];
  const float* subln  = (const float*)d_in[9];   // (128,)
  float* out = (float*)d_out;                    // (2048, 2560)

  // workspace layout (~88 MB)
  float* qkv   = (float*)d_ws;                         // 2048*3072
  float* attn1 = qkv   + (size_t)S_LEN * QKV_N;        // 2048*20*128
  float* attn2 = attn1 + (size_t)S_LEN * NHP * 128;
  float* attnm = attn2 + (size_t)S_LEN * NHP * 128;    // 2048*2560
  float* lam   = attnm + (size_t)S_LEN * HDIM;         // 1

  // 1) qkv = hs @ Wqkv^T + b
  gemm_bf16w_kernel<<<dim3(QKV_N / 128, S_LEN / 128), 256, 0, stream>>>(
      hs, Wqkv_w, Wqkv_b, qkv, S_LEN, QKV_N, HDIM);

  // 2) lambda_full
  lambda_kernel<<<1, 64, 0, stream>>>(lq1, lk1, lq2, lk2, lam);

  // 3) differential flash attention
  phi4_diff_attn_kernel<<<dim3(S_LEN / 64, NHP, 2), 128, 0, stream>>>(
      qkv, attn1, attn2);

  // 4) diff + rmsnorm
  diff_rms_kernel<<<(S_LEN * NHP) / 8, 256, 0, stream>>>(
      attn1, attn2, subln, lam, attnm);

  // 5) out = attnm @ out_w^T + out_b
  gemm_bf16w_kernel<<<dim3(HDIM / 128, S_LEN / 128), 256, 0, stream>>>(
      attnm, out_w, out_b, out, S_LEN, HDIM, HDIM);
}
